// NestedAttnProcessor2_0_72645076844886
// MI455X (gfx1250) — matrix operand order repaired and checked by reading, not compiled, not run on hardware
//
#include <hip/hip_runtime.h>
#include <hip/hip_bf16.h>

typedef __attribute__((ext_vector_type(16))) _Float16 v16h;
typedef __attribute__((ext_vector_type(8)))  _Float16 v8h;
typedef __attribute__((ext_vector_type(2)))  _Float16 v2h;
typedef __attribute__((ext_vector_type(8)))  float    v8f;

#define BATCH 2
#define HW    4096
#define HID   1280
#define CAD   2048
#define LTXT  77
#define NTOK  4
#define HEADS 20
#define HD    64

__device__ __forceinline__ v8f wmma16(v16h a, v16h b, v8f c) {
  return __builtin_amdgcn_wmma_f32_16x16x32_f16(false, a, false, b, (short)0, c,
                                                false, false);
}

// A fragment (16x32): lane(m, h) takes K = {8h..8h+7} and {16+8h..16+8h+7}
// of a row-major 32-f16 row -> two 16B loads.
__device__ __forceinline__ v16h ldfragA(const _Float16* rowp, int hh) {
  v8h lo = *(const v8h*)(rowp + hh * 8);
  v8h hi = *(const v8h*)(rowp + 16 + hh * 8);
  return __builtin_shufflevector(lo, hi, 0, 1, 2, 3, 4, 5, 6, 7,
                                 8, 9, 10, 11, 12, 13, 14, 15);
}
// B fragment (32x16): lane(n, h) takes K = {8h..8h+7} and {16+8h..16+8h+7} of column n; B is
// staged TRANSPOSED in LDS (colp = row of B^T) -> two 16B loads.
__device__ __forceinline__ v16h ldfragB(const _Float16* colp, int hh) {
  v8h lo = *(const v8h*)(colp + hh * 8);
  v8h hi = *(const v8h*)(colp + 16 + hh * 8);
  return __builtin_shufflevector(lo, hi, 0, 1, 2, 3, 4, 5, 6, 7,
                                 8, 9, 10, 11, 12, 13, 14, 15);
}

__device__ __forceinline__ v2h ld2(const float* p) {
  float2 t = *(const float2*)p;
  return (v2h){(_Float16)t.x, (_Float16)t.y};
}
__device__ __forceinline__ v2h ld2(const _Float16* p) { return *(const v2h*)p; }

// ---------------------------------------------------------------------------
__global__ void split_encoder(const float* __restrict__ enc,
                              float* __restrict__ txt, float* __restrict__ img) {
  int i = blockIdx.x * blockDim.x + threadIdx.x;
  const int total = BATCH * (LTXT + NTOK) * CAD;
  if (i >= total) return;
  int c = i % CAD;
  int r = (i / CAD) % (LTXT + NTOK);
  int b = i / (CAD * (LTXT + NTOK));
  float v = enc[i];
  if (r < LTXT) txt[(b * LTXT + r) * CAD + c] = v;
  else          img[(b * NTOK + (r - LTXT)) * CAD + c] = v;
}

// ---------------------------------------------------------------------------
// WMMA GEMM: C[M,N] = A[M,K]*B[K,N] (+bias). A f32/f16, B f32 (converted to
// f16 in LDS). 256 thr = 8 waves, block tile 64x128, K-step 32.
// A staged row-major (stride 56), B staged transposed (Bt[n][k], stride 56):
// every fragment = two ds_load_b128.
// ---------------------------------------------------------------------------
template <typename TA>
__global__ void __launch_bounds__(256)
gemm_wmma(const TA* __restrict__ A, const float* __restrict__ B,
          _Float16* __restrict__ outH, float* __restrict__ outF,
          const float* __restrict__ bias, int M, int N, int K) {
  __shared__ __align__(16) _Float16 As[64][56];    // 64 x 32 used
  __shared__ __align__(16) _Float16 Bt[128][56];   // B^T: 128(N) x 32(K) used

  const int tid  = threadIdx.x;
  const int lane = tid & 31;
  const int wave = tid >> 5;
  const int wm   = wave & 3;
  const int wn   = wave >> 2;
  const int n16  = lane & 15;
  const int hh   = lane >> 4;
  const int blockM = blockIdx.x * 64;
  const int blockN = blockIdx.y * 128;

  v8f acc[4];
#pragma unroll
  for (int t = 0; t < 4; ++t) acc[t] = (v8f){};

  for (int kb = 0; kb < K; kb += 32) {
    // stage A 64x32 as 1024 pairs (4/thread), packed v2h stores
#pragma unroll
    for (int i = 0; i < 4; ++i) {
      int e = i * 256 + tid;
      int r = e >> 4, cp = e & 15;
      int gr = blockM + r;
      v2h v = (v2h){};
      if (gr < M) v = ld2(&A[(long)gr * K + kb + 2 * cp]);
      *(v2h*)&As[r][2 * cp] = v;
    }
    // stage B transposed: 2048 pairs (8/thread); pair = B[k][n],B[k+1][n]
#pragma unroll
    for (int i = 0; i < 8; ++i) {
      int e = i * 256 + tid;
      int col = e & 127, kp = e >> 7;
      int gc = blockN + col;
      v2h v = (v2h){};
      if (gc < N) {
        float b0 = B[(long)(kb + 2 * kp) * N + gc];
        float b1 = B[(long)(kb + 2 * kp + 1) * N + gc];
        v = (v2h){(_Float16)b0, (_Float16)b1};
      }
      *(v2h*)&Bt[col][2 * kp] = v;
    }
    __syncthreads();

    v16h afr = ldfragA(&As[wm * 16 + n16][0], hh);
#pragma unroll
    for (int nt = 0; nt < 4; ++nt) {
      v16h bfr = ldfragB(&Bt[wn * 64 + nt * 16 + n16][0], hh);
      acc[nt] = wmma16(afr, bfr, acc[nt]);
    }
    __syncthreads();
  }

#pragma unroll
  for (int nt = 0; nt < 4; ++nt) {
#pragma unroll
    for (int j = 0; j < 8; ++j) {
      int row = blockM + wm * 16 + j + hh * 8;
      int col = blockN + wn * 64 + nt * 16 + n16;
      if (row < M && col < N) {
        float v = acc[nt][j];
        if (bias) v += bias[col];
        if (outF) outF[(long)row * N + col] = v;
        if (outH) outH[(long)row * N + col] = (_Float16)v;
      }
    }
  }
}

// ---------------------------------------------------------------------------
// Fused nested attention per (b, head, 64 q-rows). 128 thr = 4 waves.
// K staged natural rows (= B^T for q@k^T); V staged transposed vT[d][tok].
// Scores f32 in LDS; softmax down-converts probs to f16 IN PLACE over the
// same rows (f16 write offset 2n trails f32 read offset 4n -> safe), so PV
// A-fragments are also two-b128 loads.
// ---------------------------------------------------------------------------
__global__ void __launch_bounds__(128)
nested_attn(const _Float16* __restrict__ Q,  const _Float16* __restrict__ Kb,
            const _Float16* __restrict__ Vb, const _Float16* __restrict__ Knb,
            const _Float16* __restrict__ Vnb, const int* __restrict__ idx,
            _Float16* __restrict__ O) {
  __shared__ __align__(16) _Float16 kS[80][72];    // K rows  [tok][d], tok>=77 zero
  __shared__ __align__(16) _Float16 vT[64][104];   // V^T     [d][tok], tok>=77 zero
  __shared__ __align__(16) _Float16 knS[16][72];   // Kn rows [tok][d], tok>=4 zero
  __shared__ __align__(16) _Float16 vnT[64][48];   // Vn^T    [d][tok], tok>=4 zero
  __shared__ __align__(16) char scBuf[4][16][336]; // f32 scores (80) / f16 probs (96)
  __shared__ __align__(16) char nscBuf[4][16][64]; // f32 nscores(16) / f16 nprobs(32)
  __shared__ float psArr[4][16];

  const int tid  = threadIdx.x;
  const int lane = tid & 31;
  const int w    = tid >> 5;
  const int n16  = lane & 15;
  const int hh   = lane >> 4;
  const int chunk = blockIdx.x;
  const int head  = blockIdx.y;
  const int b     = blockIdx.z;
  const int sidx  = idx[b];
  const float scale = 0.125f;

  // ---- stage K (v8h copies) ----
  for (int e = tid; e < 80 * 8; e += 128) {
    int tok = e >> 3, dc = (e & 7) * 8;
    v8h v = (v8h){};
    if (tok < LTXT)
      v = *(const v8h*)&Kb[(long)(b * LTXT + tok) * HID + head * HD + dc];
    *(v8h*)&kS[tok][dc] = v;
  }
  // ---- stage V transposed (pairs along token) ----
  for (int e = tid; e < 48 * 64; e += 128) {
    int tp = e >> 6, d = e & 63;
    int t0 = 2 * tp, t1 = t0 + 1;
    _Float16 a = (t0 < LTXT) ? Vb[(long)(b * LTXT + t0) * HID + head * HD + d]
                             : (_Float16)0.f;
    _Float16 c = (t1 < LTXT) ? Vb[(long)(b * LTXT + t1) * HID + head * HD + d]
                             : (_Float16)0.f;
    *(v2h*)&vT[d][t0] = (v2h){a, c};
  }
  // ---- stage Kn ----
  for (int e = tid; e < 16 * 8; e += 128) {
    int tok = e >> 3, dc = (e & 7) * 8;
    v8h v = (v8h){};
    if (tok < NTOK)
      v = *(const v8h*)&Knb[(long)(b * NTOK + tok) * HID + head * HD + dc];
    *(v8h*)&knS[tok][dc] = v;
  }
  // ---- stage Vn transposed ----
  for (int e = tid; e < 16 * 64; e += 128) {
    int tp = e >> 6, d = e & 63;
    int t0 = 2 * tp, t1 = t0 + 1;
    _Float16 a = (t0 < NTOK) ? Vnb[(long)(b * NTOK + t0) * HID + head * HD + d]
                             : (_Float16)0.f;
    _Float16 c = (t1 < NTOK) ? Vnb[(long)(b * NTOK + t1) * HID + head * HD + d]
                             : (_Float16)0.f;
    *(v2h*)&vnT[d][t0] = (v2h){a, c};
  }
  __syncthreads();

  const int rowBase = chunk * 64 + w * 16;

  // ---- q fragments straight from global (16B loads) ----
  const _Float16* qrow = Q + (long)(b * HW + rowBase + n16) * HID + head * HD;
  v16h qa0 = ldfragA(qrow, hh);
  v16h qa1 = ldfragA(qrow + 32, hh);

  // ---- text scores: 5 N-tiles of 16 ----
#pragma unroll
  for (int nt = 0; nt < 5; ++nt) {
    v8f acc = (v8f){};
    acc = wmma16(qa0, ldfragB(&kS[nt * 16 + n16][0], hh), acc);
    acc = wmma16(qa1, ldfragB(&kS[nt * 16 + n16][32], hh), acc);
#pragma unroll
    for (int j = 0; j < 8; ++j)
      ((float*)scBuf[w][j + hh * 8])[nt * 16 + n16] = acc[j] * scale;
  }
  // ---- nested scores: 1 N-tile ----
  {
    v8f acc = (v8f){};
    acc = wmma16(qa0, ldfragB(&knS[n16][0], hh), acc);
    acc = wmma16(qa1, ldfragB(&knS[n16][32], hh), acc);
#pragma unroll
    for (int j = 0; j < 8; ++j)
      ((float*)nscBuf[w][j + hh * 8])[n16] = acc[j] * scale;
  }
  __syncthreads();

  // ---- softmax, one lane per row; probs written f16 in place ----
  if (lane < 16) {
    int m = lane;
    float*    srow = (float*)scBuf[w][m];
    _Float16* prow = (_Float16*)scBuf[w][m];
    float mx = -1e30f;
    for (int n = 0; n < LTXT; ++n) mx = fmaxf(mx, srow[n]);
    float sum = 0.f, es = 0.f;
    for (int n = 0; n < LTXT; ++n) {
      float e = __expf(srow[n] - mx);
      sum += e;
      if (n == sidx) es = e;
      prow[n] = (_Float16)e;     // write offset 2n trails read offset 4n
    }
    float inv = 1.f / sum;
    for (int n = 0; n < LTXT; ++n) prow[n] = (_Float16)((float)prow[n] * inv);
    for (int n = LTXT; n < 96; ++n) prow[n] = (_Float16)0.f;
    psArr[w][m] = es * inv;

    float*    nsrow = (float*)nscBuf[w][m];
    _Float16* nprow = (_Float16*)nscBuf[w][m];
    float nmx = -1e30f;
    for (int n = 0; n < NTOK; ++n) nmx = fmaxf(nmx, nsrow[n]);
    float nsum = 0.f;
    for (int n = 0; n < NTOK; ++n) {
      float e = __expf(nsrow[n] - nmx);
      nsum += e;
      nprow[n] = (_Float16)e;
    }
    float ninv = 1.f / nsum;
    for (int n = 0; n < NTOK; ++n) nprow[n] = (_Float16)((float)nprow[n] * ninv);
    for (int n = NTOK; n < 32; ++n) nprow[n] = (_Float16)0.f;
  }
  __syncthreads();

  // ---- base = P @ V (K=96), value_id = Pn @ Vn (K=32) ----
  v8f baseAcc[4], vidAcc[4];
#pragma unroll
  for (int t = 0; t < 4; ++t) { baseAcc[t] = (v8f){}; vidAcc[t] = (v8f){}; }

  const _Float16* prow = (const _Float16*)scBuf[w][n16];
#pragma unroll
  for (int ks = 0; ks < 3; ++ks) {
    v16h pa = ldfragA(prow + ks * 32, hh);
#pragma unroll
    for (int nt = 0; nt < 4; ++nt) {
      v16h bfr = ldfragB(&vT[nt * 16 + n16][ks * 32], hh);
      baseAcc[nt] = wmma16(pa, bfr, baseAcc[nt]);
    }
  }
  {
    v16h pa = ldfragA((const _Float16*)nscBuf[w][n16], hh);
#pragma unroll
    for (int nt = 0; nt < 4; ++nt) {
      v16h bfr = ldfragB(&vnT[nt * 16 + n16][0], hh);
      vidAcc[nt] = wmma16(pa, bfr, vidAcc[nt]);
    }
  }

  // ---- combine: out = base + p_s * (value_id - v_s) ----
#pragma unroll
  for (int nt = 0; nt < 4; ++nt) {
#pragma unroll
    for (int j = 0; j < 8; ++j) {
      int m = j + hh * 8;
      int d = nt * 16 + n16;
      float vsd = (float)vT[d][sidx];   // v_s[d] from transposed V
      float val = baseAcc[nt][j] + psArr[w][m] * (vidAcc[nt][j] - vsd);
      O[(long)(b * HW + rowBase + m) * HID + head * HD + d] = (_Float16)val;
    }
  }
}

// ---------------------------------------------------------------------------
extern "C" void kernel_launch(void* const* d_in, const int* in_sizes, int n_in,
                              void* d_out, int out_size, void* d_ws, size_t ws_size,
                              hipStream_t stream) {
  const float* hs   = (const float*)d_in[0];
  const float* enc  = (const float*)d_in[1];
  const int*   idx  = (const int*)d_in[2];
  const float* Wq   = (const float*)d_in[3];
  const float* Wk   = (const float*)d_in[4];
  const float* Wv   = (const float*)d_in[5];
  const float* Wkn  = (const float*)d_in[6];
  const float* Wvn  = (const float*)d_in[7];
  const float* Wout = (const float*)d_in[8];
  const float* bout = (const float*)d_in[9];
  float* out = (float*)d_out;
  (void)in_sizes; (void)n_in; (void)out_size; (void)ws_size;

  char* p = (char*)d_ws;
  auto carve = [&](size_t bytes) {
    char* r = p;
    p += (bytes + 255) & ~(size_t)255;
    return r;
  };
  float*    txt = (float*)carve((size_t)BATCH * LTXT * CAD * 4);
  float*    img = (float*)carve((size_t)BATCH * NTOK * CAD * 4);
  _Float16* Qb  = (_Float16*)carve((size_t)BATCH * HW * HID * 2);
  _Float16* Kf  = (_Float16*)carve((size_t)BATCH * LTXT * HID * 2);
  _Float16* Vf  = (_Float16*)carve((size_t)BATCH * LTXT * HID * 2);
  _Float16* Knf = (_Float16*)carve((size_t)BATCH * NTOK * HID * 2);
  _Float16* Vnf = (_Float16*)carve((size_t)BATCH * NTOK * HID * 2);
  _Float16* AO  = (_Float16*)carve((size_t)BATCH * HW * HID * 2);

  {
    int n = BATCH * (LTXT + NTOK) * CAD;
    split_encoder<<<(n + 255) / 256, 256, 0, stream>>>(enc, txt, img);
  }

  const dim3 blk(256);
  gemm_wmma<float><<<dim3((BATCH * HW + 63) / 64, HID / 128), blk, 0, stream>>>(
      hs, Wq, Qb, nullptr, nullptr, BATCH * HW, HID, HID);
  gemm_wmma<float><<<dim3((BATCH * LTXT + 63) / 64, HID / 128), blk, 0, stream>>>(
      txt, Wk, Kf, nullptr, nullptr, BATCH * LTXT, HID, CAD);
  gemm_wmma<float><<<dim3((BATCH * LTXT + 63) / 64, HID / 128), blk, 0, stream>>>(
      txt, Wv, Vf, nullptr, nullptr, BATCH * LTXT, HID, CAD);
  gemm_wmma<float><<<dim3(1, HID / 128), blk, 0, stream>>>(
      img, Wkn, Knf, nullptr, nullptr, BATCH * NTOK, HID, CAD);
  gemm_wmma<float><<<dim3(1, HID / 128), blk, 0, stream>>>(
      img, Wvn, Vnf, nullptr, nullptr, BATCH * NTOK, HID, CAD);
  nested_attn<<<dim3(HW / 64, HEADS, BATCH), dim3(128), 0, stream>>>(
      Qb, Kf, Vf, Knf, Vnf, idx, AO);
  gemm_wmma<_Float16><<<dim3((BATCH * HW + 63) / 64, HID / 128), blk, 0, stream>>>(
      AO, Wout, nullptr, out, bout, BATCH * HW, HID, HID);
}
